// LocalMultiheadAttention_9887014716268
// MI455X (gfx1250) — matrix operand order repaired and checked by reading, not compiled, not run on hardware
//
#include <hip/hip_runtime.h>
#include <stdint.h>

#define S_LEN 4096
#define BATCH 2
#define EMB   2048
#define HEADS 16
#define DHEAD 128
#define WIN   512

typedef unsigned short u16;
typedef __attribute__((ext_vector_type(16))) __bf16 bf16x16;
typedef __attribute__((ext_vector_type(8)))  __bf16 bf16x8;
typedef __attribute__((ext_vector_type(2)))  __bf16 bf16x2;
typedef __attribute__((ext_vector_type(8)))  float  f32x8;

union Frag16 {
  bf16x16 v;
  bf16x8  h[2];
};

__device__ __forceinline__ u16 f32_to_bf16(float f) {
  uint32_t u = __builtin_bit_cast(uint32_t, f);
  uint32_t r = u + 0x7FFFu + ((u >> 16) & 1u);   // round-to-nearest-even
  return (u16)(r >> 16);
}

// Packed f32x2 -> bf16x2 (single VALU op when hardware cvt is available)
__device__ __forceinline__ uint32_t pk_bf16(float a, float b) {
#if __has_builtin(__builtin_amdgcn_cvt_pk_bf16_f32)
  bf16x2 v = __builtin_amdgcn_cvt_pk_bf16_f32(a, b);
  return __builtin_bit_cast(uint32_t, v);
#else
  return (uint32_t)f32_to_bf16(a) | ((uint32_t)f32_to_bf16(b) << 16);
#endif
}

// Async global->LDS copy of 16 bytes per lane (ASYNCcnt-tracked DMA path).
// LDS byte offset = low 32 bits of the generic pointer (LDS aperture).
__device__ __forceinline__ void async_copy16(u16* lds_dst, const u16* gsrc) {
  uint32_t lds = (uint32_t)(uintptr_t)lds_dst;
  asm volatile("global_load_async_to_lds_b128 %0, %1, off"
               :: "v"(lds), "v"(gsrc) : "memory");
}
__device__ __forceinline__ void async_wait0() {
  asm volatile("s_wait_asynccnt 0x0" ::: "memory");
}

// ---------------------------------------------------------------------------
// Kernel 0: convert all four weight matrices f32 -> bf16 into workspace
// ---------------------------------------------------------------------------
__global__ __launch_bounds__(256)
void convert_weights(const float* __restrict__ Wq, const float* __restrict__ Wk,
                     const float* __restrict__ Wv, const float* __restrict__ Wo,
                     u16* __restrict__ Wb) {
  const int total = 4 * EMB * EMB;
  uint32_t* dst = (uint32_t*)Wb;
  for (int i = (blockIdx.x * blockDim.x + threadIdx.x) * 2; i < total;
       i += gridDim.x * blockDim.x * 2) {
    int j = i / (EMB * EMB);
    int r = i - j * (EMB * EMB);
    const float* src = (j == 0) ? Wq : (j == 1) ? Wk : (j == 2) ? Wv : Wo;
    float2 f = *(const float2*)(src + r);
    dst[i >> 1] = pk_bf16(f.x, f.y);
  }
}

// ---------------------------------------------------------------------------
// Kernel 1: QKV projection GEMM.  C[m, f] = sum_e x[m, e] * W[f, e]
//   x : [S,B,E] f32 (rows m = b*S + s, read permuted), W bf16 [F,E]
//   out: bf16 [B,H,S,D]   (blockIdx.z = which of q/k/v)
// ---------------------------------------------------------------------------
__global__ __launch_bounds__(256)
void qkv_proj(const float* __restrict__ xq, const float* __restrict__ xk,
              const float* __restrict__ xv, const u16* __restrict__ Wb,
              u16* __restrict__ qkv) {
  __shared__ u16 ldsA[128 * 32];   // A tile, row-major [m][k]
  __shared__ u16 ldsB[128 * 32];   // B tile transposed [n][k]

  const int which = blockIdx.z;
  const float* x = (which == 0) ? xq : (which == 1) ? xk : xv;
  const u16* W = Wb + (size_t)which * EMB * EMB;
  u16* dst = qkv + (size_t)which * BATCH * HEADS * S_LEN * DHEAD;

  const int n0 = blockIdx.x * 128;
  const int m0 = blockIdx.y * 128;
  const int b  = m0 >> 12;           // S_LEN = 4096
  const int s0 = m0 & (S_LEN - 1);

  const int tid = threadIdx.x, wv = tid >> 5, lane = tid & 31;
  const int laneN = lane & 15, hi = lane >> 4;

  f32x8 acc[8] = {};

  for (int k0 = 0; k0 < EMB; k0 += 32) {
    __syncthreads();
    // Stage B tile (bf16 weights) via async DMA: 32B per thread
    {
      int n  = tid >> 1;
      int kb = (tid & 1) * 16;
      const u16* src = W + (size_t)(n0 + n) * EMB + k0 + kb;
      async_copy16(&ldsB[n * 32 + kb], src);
      async_copy16(&ldsB[n * 32 + kb + 8], src + 8);
    }
    // Stage A: 128 rows x 32 cols, f32 -> bf16 packed converts
    for (int it = tid; it < 512; it += 256) {
      int r  = it >> 2;
      int cb = (it & 3) * 8;
      const float* src = x + (size_t)(s0 + r) * (BATCH * EMB) +
                         (size_t)b * EMB + k0 + cb;
      float4 a0 = *(const float4*)(src);
      float4 a1 = *(const float4*)(src + 4);
      uint32_t* d = (uint32_t*)&ldsA[r * 32 + cb];
      d[0] = pk_bf16(a0.x, a0.y);
      d[1] = pk_bf16(a0.z, a0.w);
      d[2] = pk_bf16(a1.x, a1.y);
      d[3] = pk_bf16(a1.z, a1.w);
    }
    async_wait0();
    __syncthreads();

    Frag16 bf;
    // B-frag (32x16): elem e -> k = e + 16*hi, n = laneN  => contiguous run
    bf.v = *(const bf16x16*)&ldsB[(wv * 16 + laneN) * 32 + (hi ? 16 : 0)];
    for (int mt = 0; mt < 8; ++mt) {
      Frag16 af;
      int row = mt * 16 + laneN;
      af.h[0] = *(const bf16x8*)&ldsA[row * 32 + (hi ? 16 : 0)];
      af.h[1] = *(const bf16x8*)&ldsA[row * 32 + 8 + (hi ? 16 : 0)];
      acc[mt] = __builtin_amdgcn_wmma_f32_16x16x32_bf16(
          false, af.v, false, bf.v, (short)0, acc[mt], false, false);
    }
  }

  // Epilogue: split columns into heads, store bf16 [B,H,S,D]
  const int f  = n0 + wv * 16 + laneN;
  const int hh = f >> 7, d = f & 127;
  for (int mt = 0; mt < 8; ++mt)
    for (int i = 0; i < 8; i += 2) {
      uint32_t pk = pk_bf16(acc[mt][i], acc[mt][i + 1]);
      int m  = m0 + mt * 16 + i + (hi ? 8 : 0);
      int bb = m >> 12, ss = m & (S_LEN - 1);
      size_t base = ((size_t)(bb * HEADS + hh) * S_LEN + ss) * DHEAD + d;
      dst[base]         = (u16)pk;
      dst[base + DHEAD] = (u16)(pk >> 16);
    }
}

// ---------------------------------------------------------------------------
// Kernel 2: local windowed causal attention (flash-style, online softmax).
// Grid: x = window*4 + rowblock (32), y = head (16), z = batch (2).
// ---------------------------------------------------------------------------
__global__ __launch_bounds__(256)
void local_attn(const u16* __restrict__ qkv, const unsigned char* __restrict__ mask,
                u16* __restrict__ attn) {
  __shared__ u16 ldsK[32 * 128];        // K chunk, [key][d]
  __shared__ u16 ldsVt[128 * 32];       // V chunk transposed, [d][key]
  __shared__ u16 ldsP[8 * 16 * 32];     // per-wave P tile, [row][col]

  const int w  = blockIdx.x >> 2;
  const int rb = blockIdx.x & 3;
  const int h  = blockIdx.y, b = blockIdx.z;
  const int tid = threadIdx.x, wv = tid >> 5, lane = tid & 31;
  const int laneN = lane & 15, hi = lane >> 4;

  const size_t bh = (size_t)(b * HEADS + h) * S_LEN;
  const size_t plane = (size_t)BATCH * HEADS * S_LEN * DHEAD;
  const u16* qg = qkv;
  const u16* kg = qkv + plane;
  const u16* vg = qkv + 2 * plane;

  const int qrow0 = w * WIN + rb * 128 + wv * 16;

  // Load Q fragments (4 x 16x32 bf16 A-frags) straight from global
  Frag16 qf[4];
  {
    const u16* qp = qg + (bh + qrow0 + laneN) * DHEAD;
    for (int dc = 0; dc < 4; ++dc) {
      qf[dc].h[0] = *(const bf16x8*)(qp + dc * 32 + (hi ? 16 : 0));
      qf[dc].h[1] = *(const bf16x8*)(qp + dc * 32 + 8 + (hi ? 16 : 0));
    }
  }

  int  qrow[8];
  bool mqb[8];
  for (int i = 0; i < 8; ++i) {
    qrow[i] = qrow0 + i + (hi ? 8 : 0);
    mqb[i]  = (mask[(size_t)b * S_LEN + qrow[i]] == 0);  // valid = !pad
  }

  f32x8 acc[8] = {};
  float mrow[8], lrow[8];
  for (int i = 0; i < 8; ++i) { mrow[i] = -3.0e38f; lrow[i] = 0.f; }

  const int kbase0 = (w - 1) * WIN;                 // key index of column 0
  const int c0 = (w == 0) ? (WIN / 32) : 0;         // skip left padding
  const int cN = (WIN + rb * 128 + 128) / 32;       // causal upper bound
  // NOTE: for ch in [c0, cN), kidx = kbase0 + ch*32 + r is always in [0, S).

  for (int ch = c0; ch < cN; ++ch) {
    __syncthreads();
    // --- stage K chunk [32][128] via async DMA (in-range by construction) --
    for (int it = tid; it < 512; it += 256) {
      int r  = it >> 4;
      int cb = (it & 15) * 8;
      int kidx = kbase0 + ch * 32 + r;
      async_copy16(&ldsK[r * 128 + cb], kg + (bh + kidx) * DHEAD + cb);
      if (ch + 1 < cN)
        __builtin_prefetch(kg + (bh + kidx + 32) * DHEAD + cb, 0, 3);
    }
    // --- stage V chunk transposed [128][32] -------------------------------
    for (int it = tid; it < 512; it += 256) {
      int r  = it >> 4;
      int cb = (it & 15) * 8;
      int kidx = kbase0 + ch * 32 + r;
      uint4 val = *(const uint4*)(vg + (bh + kidx) * DHEAD + cb);
      union { uint4 q; u16 u[8]; } tv; tv.q = val;
      for (int j = 0; j < 8; ++j) ldsVt[(cb + j) * 32 + r] = tv.u[j];
      if (ch + 1 < cN)
        __builtin_prefetch(vg + (bh + kidx + 32) * DHEAD + cb, 0, 3);
    }
    async_wait0();
    __syncthreads();

    // --- QK^T: two 16-column score tiles, K-depth 128 = 4 WMMAs each ------
    f32x8 s0 = {}, s1 = {};
    for (int dc = 0; dc < 4; ++dc) {
      bf16x16 b0 = *(const bf16x16*)&ldsK[(laneN) * 128 + dc * 32 + (hi ? 16 : 0)];
      bf16x16 b1 = *(const bf16x16*)&ldsK[(16 + laneN) * 128 + dc * 32 + (hi ? 16 : 0)];
      s0 = __builtin_amdgcn_wmma_f32_16x16x32_bf16(false, qf[dc].v, false, b0,
                                                   (short)0, s0, false, false);
      s1 = __builtin_amdgcn_wmma_f32_16x16x32_bf16(false, qf[dc].v, false, b1,
                                                   (short)0, s1, false, false);
    }

    // --- masking + online softmax ----------------------------------------
    const float scale = 0.0883883476483184f;   // 1/sqrt(128)
    const int k0i = kbase0 + ch * 32 + laneN;
    const int k1i = k0i + 16;
    bool v0 = (mask[(size_t)b * S_LEN + k0i] == 0);
    bool v1 = (mask[(size_t)b * S_LEN + k1i] == 0);

    float cf[8];
    for (int i = 0; i < 8; ++i) {
      bool ok0 = v0 && mqb[i] && (k0i <= qrow[i]);
      bool ok1 = v1 && mqb[i] && (k1i <= qrow[i]);
      float t0 = ok0 ? s0[i] * scale : -3.0e38f;
      float t1 = ok1 ? s1[i] * scale : -3.0e38f;
      // row max across the 16 lanes holding this row's columns
      float tm = fmaxf(t0, t1);
      for (int m = 1; m < 16; m <<= 1) tm = fmaxf(tm, __shfl_xor(tm, m, 32));
      float newm = fmaxf(mrow[i], tm);
      cf[i] = __expf(mrow[i] - newm);
      mrow[i] = newm;
      float p0 = (t0 > -1e37f) ? __expf(t0 - newm) : 0.f;
      float p1 = (t1 > -1e37f) ? __expf(t1 - newm) : 0.f;
      float rs = p0 + p1;
      for (int m = 1; m < 16; m <<= 1) rs += __shfl_xor(rs, m, 32);
      lrow[i] = lrow[i] * cf[i] + rs;
      int row = i + (hi ? 8 : 0);
      uint32_t pk = pk_bf16(p0, p1);
      ldsP[wv * 512 + row * 32 + laneN]      = (u16)pk;
      ldsP[wv * 512 + row * 32 + 16 + laneN] = (u16)(pk >> 16);
    }
    for (int nt = 0; nt < 8; ++nt)
      for (int i = 0; i < 8; ++i) acc[nt][i] *= cf[i];

    // --- P (C-layout) -> A-fragment via per-wave LDS, then P @ V ----------
    Frag16 pf;
    pf.h[0] = *(const bf16x8*)&ldsP[wv * 512 + laneN * 32 + (hi ? 16 : 0)];
    pf.h[1] = *(const bf16x8*)&ldsP[wv * 512 + laneN * 32 + 8 + (hi ? 16 : 0)];
    for (int nt = 0; nt < 8; ++nt) {
      bf16x16 vf = *(const bf16x16*)&ldsVt[(nt * 16 + laneN) * 32 + (hi ? 16 : 0)];
      acc[nt] = __builtin_amdgcn_wmma_f32_16x16x32_bf16(
          false, pf.v, false, vf, (short)0, acc[nt], false, false);
    }
  }

  // Epilogue: normalize and store bf16 attn_out [B,S,E] (heads re-merged)
  float rinv[8];
  for (int i = 0; i < 8; ++i) rinv[i] = (lrow[i] > 0.f) ? 1.f / lrow[i] : 0.f;
  for (int nt = 0; nt < 8; ++nt)
    for (int i = 0; i < 8; i += 2) {
      uint32_t pk = pk_bf16(acc[nt][i] * rinv[i], acc[nt][i + 1] * rinv[i + 1]);
      size_t base = ((size_t)b * S_LEN + qrow[i]) * EMB + h * DHEAD + nt * 16 + laneN;
      attn[base]       = (u16)pk;
      attn[base + EMB] = (u16)(pk >> 16);
    }
}

// ---------------------------------------------------------------------------
// Kernel 3: output projection GEMM + bias, f32 epilogue to [S,B,E]
// ---------------------------------------------------------------------------
__global__ __launch_bounds__(256)
void out_proj(const u16* __restrict__ attn, const u16* __restrict__ Wo,
              const float* __restrict__ bo, float* __restrict__ out) {
  __shared__ u16 ldsA[128 * 32];
  __shared__ u16 ldsB[128 * 32];

  const int n0 = blockIdx.x * 128;
  const int m0 = blockIdx.y * 128;
  const int tid = threadIdx.x, wv = tid >> 5, lane = tid & 31;
  const int laneN = lane & 15, hi = lane >> 4;

  f32x8 acc[8] = {};

  for (int k0 = 0; k0 < EMB; k0 += 32) {
    __syncthreads();
    {
      int r  = tid >> 1;
      int kb = (tid & 1) * 16;
      const u16* sa = attn + (size_t)(m0 + r) * EMB + k0 + kb;
      async_copy16(&ldsA[r * 32 + kb], sa);
      async_copy16(&ldsA[r * 32 + kb + 8], sa + 8);
      const u16* sb = Wo + (size_t)(n0 + r) * EMB + k0 + kb;
      async_copy16(&ldsB[r * 32 + kb], sb);
      async_copy16(&ldsB[r * 32 + kb + 8], sb + 8);
    }
    async_wait0();
    __syncthreads();

    Frag16 bf;
    bf.v = *(const bf16x16*)&ldsB[(wv * 16 + laneN) * 32 + (hi ? 16 : 0)];
    for (int mt = 0; mt < 8; ++mt) {
      Frag16 af;
      int row = mt * 16 + laneN;
      af.h[0] = *(const bf16x8*)&ldsA[row * 32 + (hi ? 16 : 0)];
      af.h[1] = *(const bf16x8*)&ldsA[row * 32 + 8 + (hi ? 16 : 0)];
      acc[mt] = __builtin_amdgcn_wmma_f32_16x16x32_bf16(
          false, af.v, false, bf.v, (short)0, acc[mt], false, false);
    }
  }

  const int f = n0 + wv * 16 + laneN;
  const float bof = bo[f];
  for (int mt = 0; mt < 8; ++mt)
    for (int i = 0; i < 8; ++i) {
      int m  = m0 + mt * 16 + i + (hi ? 8 : 0);
      int bb = m >> 12, ss = m & (S_LEN - 1);
      out[((size_t)ss * BATCH + bb) * EMB + f] = acc[mt][i] + bof;
    }
}

// ---------------------------------------------------------------------------
extern "C" void kernel_launch(void* const* d_in, const int* in_sizes, int n_in,
                              void* d_out, int out_size, void* d_ws, size_t ws_size,
                              hipStream_t stream) {
  (void)in_sizes; (void)n_in; (void)out_size; (void)ws_size;
  const float* q  = (const float*)d_in[0];
  const float* k  = (const float*)d_in[1];
  const float* v  = (const float*)d_in[2];
  const unsigned char* mask = (const unsigned char*)d_in[3];
  const float* Wq = (const float*)d_in[4];
  const float* Wk = (const float*)d_in[5];
  const float* Wv = (const float*)d_in[6];
  const float* Wo = (const float*)d_in[7];
  const float* bo = (const float*)d_in[8];

  u16* ws   = (u16*)d_ws;
  u16* Wb   = ws;                                           // 4*E*E bf16
  u16* qkv  = Wb + (size_t)4 * EMB * EMB;                   // 3*[B,H,S,D] bf16
  u16* attn = qkv + (size_t)3 * BATCH * HEADS * S_LEN * DHEAD; // [B,S,E] bf16

  convert_weights<<<1024, 256, 0, stream>>>(Wq, Wk, Wv, Wo, Wb);
  qkv_proj<<<dim3(16, 64, 3), 256, 0, stream>>>(q, k, v, Wb, qkv);
  local_attn<<<dim3(32, 16, 2), 256, 0, stream>>>(qkv, mask, attn);
  out_proj<<<dim3(16, 64), 256, 0, stream>>>(attn, Wb + (size_t)3 * EMB * EMB, bo,
                                             (float*)d_out);
}
